// GCN_1382979470185
// MI455X (gfx1250) — matrix operand order repaired and checked by reading, not compiled, not run on hardware
//
#include <hip/hip_runtime.h>
#include <hip/hip_bf16.h>
#include <stdint.h>

#define N_NODES   10000
#define N_EDGES   320000
#define IN_FEATS  256
#define HIDDEN    256
#define N_CLASSES 64

typedef __attribute__((ext_vector_type(16))) __bf16 v16bf;
typedef __attribute__((ext_vector_type(8)))  float  v8f;

union BFrag {
    v16bf v;
    unsigned short s[16];
    uint4 q[2];
};

__device__ __forceinline__ unsigned short f2bf(float f) {
    unsigned u = __float_as_uint(f);
    u += 0x7FFFu + ((u >> 16) & 1u);   // round-to-nearest-even
    return (unsigned short)(u >> 16);
}

// ---------------- setup kernels ----------------

__global__ void k_zero(float* p, int n) {
    int i = blockIdx.x * blockDim.x + threadIdx.x;
    int stride = gridDim.x * blockDim.x;
    for (; i < n; i += stride) p[i] = 0.0f;
}

__global__ void k_deg(const int* __restrict__ src, const int* __restrict__ dst,
                      float* out_deg, float* in_deg) {
    int e = blockIdx.x * blockDim.x + threadIdx.x;
    if (e < N_EDGES) {
        atomicAdd(&out_deg[src[e]], 1.0f);
        atomicAdd(&in_deg[dst[e]], 1.0f);
    }
}

__global__ void k_norm(const float* __restrict__ out_deg, const float* __restrict__ in_deg,
                       float* norm_src, float* norm_dst) {
    int i = blockIdx.x * blockDim.x + threadIdx.x;
    if (i < N_NODES) {
        norm_src[i] = rsqrtf(fmaxf(out_deg[i], 1.0f));
        norm_dst[i] = rsqrtf(fmaxf(in_deg[i], 1.0f));
    }
}

// agg1[v][f] += features[u][f] * norm_src[u]  for each edge u->v. One block per edge.
__global__ void k_agg1(const float* __restrict__ feat, const int* __restrict__ src,
                       const int* __restrict__ dst, const float* __restrict__ norm_src,
                       float* agg) {
    int e = blockIdx.x;
    int f = threadIdx.x;
    int u = src[e], v = dst[e];
    float val = feat[(size_t)u * IN_FEATS + f] * norm_src[u];
    atomicAdd(&agg[(size_t)v * IN_FEATS + f], val);
}

// A-side precision/scale pass: aggbf[i] = bf16(agg1[i] * norm_dst[row])
__global__ void k_cvtA(const float* __restrict__ agg1, const float* __restrict__ norm_dst,
                       unsigned short* __restrict__ aggbf) {
    int i = blockIdx.x * blockDim.x + threadIdx.x;
    if (i < N_NODES * IN_FEATS) {
        int row = i >> 8;   // IN_FEATS == 256
        aggbf[i] = f2bf(agg1[i] * norm_dst[row]);
    }
}

// Pack KxN f32 weight into bf16 WMMA B-fragment layout.
// Fragment (kt,nt): 512 bf16 = [lane 0..31][j 0..15],
//   element = W[kt*32 + (lane>=16 ? 8:0) + (j<8 ? j : j+8)][nt*16 + (lane&15)]
__global__ void k_pack(const float* __restrict__ W, unsigned short* Wp, int K, int N) {
    int tid = blockIdx.x * blockDim.x + threadIdx.x;
    int NT = N / 16;
    int total = (K / 32) * NT * 512;
    if (tid >= total) return;
    int frag = tid >> 9;
    int within = tid & 511;
    int lane = within >> 4;
    int j = within & 15;
    int kt = frag / NT;
    int nt = frag - kt * NT;
    int k = kt * 32 + ((lane >= 16) ? 8 : 0) + ((j < 8) ? j : j + 8);
    int n = nt * 16 + (lane & 15);
    Wp[tid] = f2bf(W[(size_t)k * N + n]);
}

// ---------------- fused GEMM1(+bias,ReLU) -> GEMM2 ----------------
// Block = 16-row strip of nodes, 4 waves. GEMM1: each wave does 4 N-tiles of 256.
// H strip staged in LDS (bf16), then GEMM2: each wave does 1 N-tile of 64.
// A-matrix (16-bit 16x32) lane layout (ISA 7.12.2):
//   lanes 0-15 : row M=lane,    K=[0..7] in s[0..7],  K=[16..23] in s[8..15]
//   lanes 16-31: row M=lane-16, K=[8..15] in s[0..7], K=[24..31] in s[8..15]
__global__ __launch_bounds__(128) void k_gemm_fused(
    const unsigned short* __restrict__ aggbf, const float* __restrict__ b1,
    const unsigned short* __restrict__ W1p, const unsigned short* __restrict__ W2p,
    float* __restrict__ Y2)
{
    __shared__ unsigned short hstrip[16 * HIDDEN];   // 8 KB bf16

    const int lane = threadIdx.x & 31;
    const int wave = threadIdx.x >> 5;
    const int row0 = blockIdx.x * 16;
    const int rloc = lane & 15;
    const int hi   = (lane >= 16) ? 1 : 0;

    v8f acc[4];
#pragma unroll
    for (int t = 0; t < 4; ++t)
#pragma unroll
        for (int r = 0; r < 8; ++r) acc[t][r] = 0.0f;

    const unsigned short* arow = aggbf + (size_t)(row0 + rloc) * IN_FEATS + hi * 8;

#pragma unroll
    for (int kt = 0; kt < 8; ++kt) {
        // A fragment: two contiguous 16B loads of pre-converted bf16
        BFrag a;
        const unsigned short* ap = arow + kt * 32;
        a.q[0] = *(const uint4*)(ap);
        a.q[1] = *(const uint4*)(ap + 16);

        if (kt < 7) {   // prefetch next K-step's B fragments
            __builtin_prefetch(W1p + (((size_t)(kt + 1) * 16 + wave * 4) * 512 + lane * 16), 0, 0);
        }

#pragma unroll
        for (int t = 0; t < 4; ++t) {
            int nt = wave * 4 + t;
            BFrag b;
            const uint4* bp = (const uint4*)(W1p + (((size_t)kt * 16 + nt) * 512 + lane * 16));
            b.q[0] = bp[0];
            b.q[1] = bp[1];
            acc[t] = __builtin_amdgcn_wmma_f32_16x16x32_bf16(
                false, a.v, false, b.v, (short)0, acc[t], false, false);
        }
    }

    // bias + ReLU, stage H strip as bf16 in LDS.
    // C layout: lane 0-15 N=lane, M=r; lanes 16-31 N=lane-16, M=r+8.
#pragma unroll
    for (int t = 0; t < 4; ++t) {
        int col = (wave * 4 + t) * 16 + rloc;
        float bias = b1[col];
#pragma unroll
        for (int r = 0; r < 8; ++r) {
            float v = fmaxf(acc[t][r] + bias, 0.0f);
            hstrip[(r + hi * 8) * HIDDEN + col] = f2bf(v);
        }
    }
    __syncthreads();

    // GEMM2: Y2_strip = H_strip @ W2, each wave owns N-tile = wave (N=64 -> 4 tiles).
    v8f acc2;
#pragma unroll
    for (int r = 0; r < 8; ++r) acc2[r] = 0.0f;

    const unsigned short* hrow = hstrip + rloc * HIDDEN + hi * 8;

#pragma unroll
    for (int kt = 0; kt < 8; ++kt) {
        BFrag a;
        const unsigned short* ap = hrow + kt * 32;
        a.q[0] = *(const uint4*)(ap);        // ds_load_b128
        a.q[1] = *(const uint4*)(ap + 16);
        BFrag b;
        const uint4* bp = (const uint4*)(W2p + (((size_t)kt * 4 + wave) * 512 + lane * 16));
        b.q[0] = bp[0];
        b.q[1] = bp[1];
        acc2 = __builtin_amdgcn_wmma_f32_16x16x32_bf16(
            false, a.v, false, b.v, (short)0, acc2, false, false);
    }

    int coln = wave * 16 + rloc;
#pragma unroll
    for (int r = 0; r < 8; ++r) {
        int m = r + hi * 8;
        Y2[(size_t)(row0 + m) * N_CLASSES + coln] = acc2[r];
    }
}

// out_acc[v][f] += Y2[u][f] * norm_src[u].  4 edges per 256-thread block.
__global__ void k_agg2(const float* __restrict__ Y2, const int* __restrict__ src,
                       const int* __restrict__ dst, const float* __restrict__ norm_src,
                       float* out_acc) {
    int e = blockIdx.x * 4 + (threadIdx.x >> 6);
    int f = threadIdx.x & 63;
    int u = src[e], v = dst[e];
    atomicAdd(&out_acc[(size_t)v * N_CLASSES + f],
              Y2[(size_t)u * N_CLASSES + f] * norm_src[u]);
}

__global__ void k_final(const float* __restrict__ out_acc, const float* __restrict__ norm_dst,
                        const float* __restrict__ b2, float* __restrict__ out) {
    int i = blockIdx.x * blockDim.x + threadIdx.x;
    if (i < N_NODES * N_CLASSES) {
        int v = i >> 6;
        int f = i & 63;
        out[i] = out_acc[i] * norm_dst[v] + b2[f];
    }
}

// ---------------- launcher ----------------

extern "C" void kernel_launch(void* const* d_in, const int* in_sizes, int n_in,
                              void* d_out, int out_size, void* d_ws, size_t ws_size,
                              hipStream_t stream) {
    const float* features = (const float*)d_in[0];
    const float* W1       = (const float*)d_in[1];
    const float* b1       = (const float*)d_in[2];
    const float* W2       = (const float*)d_in[3];
    const float* b2       = (const float*)d_in[4];
    const int*   src      = (const int*)d_in[5];
    const int*   dst      = (const int*)d_in[6];
    float* out = (float*)d_out;

    // workspace layout (float units)
    float* ws       = (float*)d_ws;
    float* out_deg  = ws;                       // 10000
    float* in_deg   = ws + 10000;               // 10000
    float* norm_src = ws + 20000;               // 10000
    float* norm_dst = ws + 30000;               // 10000
    float* agg1     = ws + 40000;               // 2,560,000 f32 (dead after k_cvtA)
    float* Y2       = agg1;                     // reuse: 640,000 f32
    float* out_acc  = agg1 + 640000;            // reuse: 640,000 f32 (re-zeroed)
    unsigned short* W1p   = (unsigned short*)(ws + 2600000);   // 65536 u16
    unsigned short* W2p   = W1p + 65536;                        // 16384 u16
    unsigned short* aggbf = (unsigned short*)(ws + 2640960);   // 2,560,000 u16

    // zero degrees + agg1
    k_zero<<<2048, 256, 0, stream>>>(ws, 2600000);

    k_deg<<<(N_EDGES + 255) / 256, 256, 0, stream>>>(src, dst, out_deg, in_deg);
    k_norm<<<(N_NODES + 255) / 256, 256, 0, stream>>>(out_deg, in_deg, norm_src, norm_dst);

    k_agg1<<<N_EDGES, 256, 0, stream>>>(features, src, dst, norm_src, agg1);
    k_cvtA<<<(N_NODES * IN_FEATS + 255) / 256, 256, 0, stream>>>(agg1, norm_dst, aggbf);

    k_pack<<<(65536 + 255) / 256, 256, 0, stream>>>(W1, W1p, IN_FEATS, HIDDEN);
    k_pack<<<(16384 + 255) / 256, 256, 0, stream>>>(W2, W2p, HIDDEN, N_CLASSES);

    k_gemm_fused<<<N_NODES / 16, 128, 0, stream>>>(aggbf, b1, W1p, W2p, Y2);

    // re-zero accumulator region then aggregate layer 2
    k_zero<<<1024, 256, 0, stream>>>(out_acc, N_NODES * N_CLASSES);
    k_agg2<<<N_EDGES / 4, 256, 0, stream>>>(Y2, src, dst, norm_src, out_acc);

    k_final<<<(N_NODES * N_CLASSES + 255) / 256, 256, 0, stream>>>(out_acc, norm_dst, b2, out);
}
